// GNN_35983236006067
// MI455X (gfx1250) — matrix operand order repaired and checked by reading, not compiled, not run on hardware
//
#include <hip/hip_runtime.h>

typedef __attribute__((ext_vector_type(16))) _Float16 v16h;
typedef __attribute__((ext_vector_type(8)))  float    v8f;

#define N_NODES 100000
#define D 128
#define N_EDGES 1600000
#define BN_EPS 1e-5f
#define SLOPE 0.01f

union Frag16 { v16h v; uint4 q[2]; };

// ---------------------------------------------------------------------------
// K1: convert W and rs_W (f32, [K=128][N=128] row-major) into f16 transposed
//     layout Wt[n*128 + k] so B fragments are contiguous per lane.
// ---------------------------------------------------------------------------
__global__ void gnn_wconv(const float* __restrict__ W, const float* __restrict__ rsW,
                          _Float16* __restrict__ Wt, _Float16* __restrict__ rsWt) {
    int t = blockIdx.x * blockDim.x + threadIdx.x;   // 0..16383
    int k = t >> 7, n = t & 127;
    Wt[n * D + k]   = (_Float16)W[k * D + n];
    rsWt[n * D + k] = (_Float16)rsW[k * D + n];
}

// ---------------------------------------------------------------------------
// K2: deg[i] = 1.0 (self loop)
// ---------------------------------------------------------------------------
__global__ void gnn_initdeg(float* __restrict__ deg) {
    int t = blockIdx.x * blockDim.x + threadIdx.x;
    if (t < N_NODES) deg[t] = 1.0f;
}

// ---------------------------------------------------------------------------
// K3: deg[dst] += 1 per edge (global_atomic_add_f32, no return)
// ---------------------------------------------------------------------------
__global__ void gnn_degacc(const int* __restrict__ ei, float* __restrict__ deg) {
    int e = blockIdx.x * blockDim.x + threadIdx.x;
    if (e < N_EDGES) atomicAdd(&deg[ei[N_EDGES + e]], 1.0f);
}

// ---------------------------------------------------------------------------
// K4: per-row L2 normalize + BatchNorm (eval), write f16 xh; dinv = rsqrt(deg)
//     One wave (32 lanes) per row; lane handles 4 features (float4).
// ---------------------------------------------------------------------------
__global__ __launch_bounds__(256) void gnn_normbn(
    const float* __restrict__ x, const float* __restrict__ gamma,
    const float* __restrict__ beta, const float* __restrict__ mean,
    const float* __restrict__ var, float* __restrict__ deg_dinv,
    _Float16* __restrict__ xh) {
    int wave = threadIdx.x >> 5, lane = threadIdx.x & 31;
    int row = blockIdx.x * 8 + wave;
    if (row >= N_NODES) return;
    int f = lane * 4;
    const float4 xv = *(const float4*)(x + (size_t)row * D + f);
    float s = xv.x * xv.x + xv.y * xv.y + xv.z * xv.z + xv.w * xv.w;
    #pragma unroll
    for (int o = 16; o >= 1; o >>= 1) s += __shfl_xor(s, o, 32);
    float inv = 1.0f / fmaxf(sqrtf(s), 1e-12f);
    const float4 g = *(const float4*)(gamma + f);
    const float4 bt = *(const float4*)(beta + f);
    const float4 mn = *(const float4*)(mean + f);
    const float4 vr = *(const float4*)(var + f);
    float o0 = (xv.x * inv - mn.x) * rsqrtf(vr.x + BN_EPS) * g.x + bt.x;
    float o1 = (xv.y * inv - mn.y) * rsqrtf(vr.y + BN_EPS) * g.y + bt.y;
    float o2 = (xv.z * inv - mn.z) * rsqrtf(vr.z + BN_EPS) * g.z + bt.z;
    float o3 = (xv.w * inv - mn.w) * rsqrtf(vr.w + BN_EPS) * g.w + bt.w;
    struct H4 { _Float16 a, b, c, d; };
    H4 h = { (_Float16)o0, (_Float16)o1, (_Float16)o2, (_Float16)o3 };
    *(H4*)(xh + (size_t)row * D + f) = h;
    if (lane == 0) deg_dinv[row] = rsqrtf(deg_dinv[row]);   // deg -> dinv in place
}

// ---------------------------------------------------------------------------
// K5: dual WMMA GEMM.  Block = 256 thr = 8 waves; block owns 16 rows (m0),
//     wave w owns N-tile n0 = w*16.  4 K-steps of v_wmma_f32_16x16x32_f16
//     with two accumulators (W path -> xw, rs_W path -> d_out + biases).
//  A as loaded here (16-bit A 16x32): lane half (lane>>4) selects K 0-15 vs
//  16-31 for row m = lane&15.  B layout: lane half selects K 0-15 vs
//  16-31 for column n = lane&15 (contiguous in the transposed Wt buffer).
// ---------------------------------------------------------------------------
__global__ __launch_bounds__(256) void gnn_gemm(
    const _Float16* __restrict__ xh, const _Float16* __restrict__ Wt,
    const _Float16* __restrict__ rsWt, const float* __restrict__ b,
    const float* __restrict__ rsb, float* __restrict__ xw,
    float* __restrict__ out) {
    int lane = threadIdx.x & 31;
    int wv   = threadIdx.x >> 5;       // 0..7
    int n0   = wv * 16;
    int m0   = blockIdx.x * 16;        // N_NODES % 16 == 0
    int which = lane >> 4;             // half-wave select
    int mr    = lane & 15;             // row (A) / column (B)

    const _Float16* arow = xh   + (size_t)(m0 + mr) * D;
    const _Float16* bW   = Wt   + (size_t)(n0 + mr) * D;
    const _Float16* bR   = rsWt + (size_t)(n0 + mr) * D;

    v8f cW = {}; v8f cR = {};
    #pragma unroll
    for (int ks = 0; ks < 4; ++ks) {
        int k0 = ks * 32;
        Frag16 fa, fbW, fbR;
        fa.q[0]  = *(const uint4*)(arow + k0 + which * 16);       // K k0+h16..+7
        fa.q[1]  = *(const uint4*)(arow + k0 + which * 16 + 8);   // K k0+h16+8..+15
        fbW.q[0] = *(const uint4*)(bW + k0 + which * 16);
        fbW.q[1] = *(const uint4*)(bW + k0 + which * 16 + 8);
        fbR.q[0] = *(const uint4*)(bR + k0 + which * 16);
        fbR.q[1] = *(const uint4*)(bR + k0 + which * 16 + 8);
        cW = __builtin_amdgcn_wmma_f32_16x16x32_f16(false, fa.v, false, fbW.v,
                                                    (short)0, cW, false, false);
        cR = __builtin_amdgcn_wmma_f32_16x16x32_f16(false, fa.v, false, fbR.v,
                                                    (short)0, cR, false, false);
    }
    int col = n0 + mr;
    float bias = b[col] + rsb[col];
    #pragma unroll
    for (int r = 0; r < 8; ++r) {     // D VGPR r -> row m0 + r + 8*which
        size_t row = (size_t)(m0 + r + which * 8);
        xw[row * D + col]  = cW[r];
        out[row * D + col] = cR[r] + bias;
    }
}

// ---------------------------------------------------------------------------
// K6: edge scatter.  One wave per edge; lane gathers float4 of xw[src] and
//     does 4 global f32 atomic adds into out[dst].
// ---------------------------------------------------------------------------
__global__ __launch_bounds__(256) void gnn_edges(
    const int* __restrict__ ei, const float* __restrict__ xw,
    const float* __restrict__ dinv, float* __restrict__ out) {
    int gid = blockIdx.x * blockDim.x + threadIdx.x;  // E*32 = 51.2M < 2^31
    int e = gid >> 5, lane = gid & 31;
    if (e >= N_EDGES) return;
    int src = ei[e];
    int dst = ei[N_EDGES + e];
    float w = dinv[src] * dinv[dst];
    const float4 v = *(const float4*)(xw + (size_t)src * D + lane * 4);
    float* op = out + (size_t)dst * D + lane * 4;
    atomicAdd(op + 0, v.x * w);
    atomicAdd(op + 1, v.y * w);
    atomicAdd(op + 2, v.z * w);
    atomicAdd(op + 3, v.w * w);
}

// ---------------------------------------------------------------------------
// K7: self-loop message + LeakyReLU epilogue.
// ---------------------------------------------------------------------------
__global__ void gnn_final(const float* __restrict__ xw,
                          const float* __restrict__ dinv,
                          float* __restrict__ out) {
    int t = blockIdx.x * blockDim.x + threadIdx.x;   // N*D = 12.8M
    if (t >= N_NODES * D) return;
    int i = t >> 7;
    float dv = dinv[i];
    float v = out[t] + xw[t] * dv * dv;
    out[t] = v >= 0.0f ? v : SLOPE * v;
}

// ---------------------------------------------------------------------------
extern "C" void kernel_launch(void* const* d_in, const int* in_sizes, int n_in,
                              void* d_out, int out_size, void* d_ws, size_t ws_size,
                              hipStream_t stream) {
    const float* x     = (const float*)d_in[0];
    const int*   ei    = (const int*)  d_in[1];
    const float* W     = (const float*)d_in[2];
    const float* b     = (const float*)d_in[3];
    const float* gamma = (const float*)d_in[4];
    const float* beta  = (const float*)d_in[5];
    const float* mean  = (const float*)d_in[6];
    const float* var   = (const float*)d_in[7];
    const float* rsW   = (const float*)d_in[8];
    const float* rsb   = (const float*)d_in[9];
    float* out = (float*)d_out;

    // workspace layout (bytes), all 16B-aligned
    char* ws = (char*)d_ws;
    _Float16* xh   = (_Float16*)(ws);                       // 25,600,000 B
    float*    xw   = (float*)   (ws + 25600000);            // 51,200,000 B
    float*    dinv = (float*)   (ws + 76800000);            //    400,000 B
    _Float16* Wt   = (_Float16*)(ws + 77200000);            //     32,768 B
    _Float16* rsWt = (_Float16*)(ws + 77232768);            //     32,768 B

    gnn_wconv  <<<64,     256, 0, stream>>>(W, rsW, Wt, rsWt);
    gnn_initdeg<<<391,    256, 0, stream>>>(dinv);
    gnn_degacc <<<6250,   256, 0, stream>>>(ei, dinv);
    gnn_normbn <<<12500,  256, 0, stream>>>(x, gamma, beta, mean, var, dinv, xh);
    gnn_gemm   <<<6250,   256, 0, stream>>>(xh, Wt, rsWt, b, rsb, xw, out);
    gnn_edges  <<<200000, 256, 0, stream>>>(ei, xw, dinv, out);
    gnn_final  <<<50000,  256, 0, stream>>>(xw, dinv, out);
}
